// Exercise_KC_GraphAttentionLayer_26414048870432
// MI455X (gfx1250) — matrix operand order repaired and checked by reading, not compiled, not run on hardware
//
#include <hip/hip_runtime.h>
#include <hip/hip_bf16.h>
#include <cstdint>

typedef __attribute__((ext_vector_type(16))) _Float16 v16h;
typedef __attribute__((ext_vector_type(8)))  float    v8f;
typedef __attribute__((ext_vector_type(2)))  float    v2f;
typedef __attribute__((ext_vector_type(4)))  float    f4;
typedef __attribute__((ext_vector_type(4)))  int      i4;
typedef __attribute__((ext_vector_type(2)))  unsigned u2;

#define N_EX   20000
#define N_KC   1024
#define F_IN   128
#define F_OUT  128
#define LEAKY  0.2f
#define MASKV  (-9000000000000000.0f)

// ---------------------------------------------------------------------------
// K0: w1a[k]      = sum_f W1[k,f]*a1[f]   (ex_Wh@a1 == ex_h@w1a1)
//     w1a[128+k]  = sum_f W1[k,f]*a2[f]   (kc_Wh@a2 == kc_h@w1a2)
// ---------------------------------------------------------------------------
__global__ void k_w1a(const float* __restrict__ W1, const float* __restrict__ a,
                      float* __restrict__ w1a) {
    int k = threadIdx.x;
    float s1 = 0.f, s2 = 0.f;
    for (int f = 0; f < F_OUT; ++f) {
        float w = W1[k * F_OUT + f];
        s1 += w * a[f];
        s2 += w * a[F_OUT + f];
    }
    w1a[k]        = s1;
    w1a[F_IN + k] = s2;
}

// ---------------------------------------------------------------------------
// K1: kc_whT[f][j] = (f16)(kc_h[j,:] . W1[:,f])   (K-major for WMMA B loads)
// ---------------------------------------------------------------------------
__global__ void k_kcwh(const float* __restrict__ kc_h, const float* __restrict__ W1,
                       _Float16* __restrict__ kc_whT) {
    int j = blockIdx.x;
    int f = threadIdx.x;
    const float* kr = kc_h + (size_t)j * F_IN;
    float acc = 0.f;
    for (int k = 0; k < F_IN; ++k)
        acc += kr[k] * W1[k * F_OUT + f];
    kc_whT[(size_t)f * N_KC + j] = (_Float16)acc;
}

// ---------------------------------------------------------------------------
// K2: attention scores, one wave32 per row (128-dot + shuffle reduce)
// ---------------------------------------------------------------------------
__global__ void k_scores(const float* __restrict__ ex_h, const float* __restrict__ kc_h,
                         const float* __restrict__ w1a,
                         float* __restrict__ ex_score, float* __restrict__ kc_score) {
    int wave = (int)((blockIdx.x * blockDim.x + threadIdx.x) >> 5);
    int lane = threadIdx.x & 31;
    if (wave >= N_EX + N_KC) return;
    const float* src; const float* vec; float* dst; int idx;
    if (wave < N_EX) { idx = wave;        src = ex_h + (size_t)idx * F_IN; vec = w1a;        dst = ex_score; }
    else             { idx = wave - N_EX; src = kc_h + (size_t)idx * F_IN; vec = w1a + F_IN; dst = kc_score; }
    float s = 0.f;
    for (int t = 0; t < 4; ++t) { int k = lane + t * 32; s += src[k] * vec[k]; }
    for (int o = 16; o > 0; o >>= 1) s += __shfl_xor(s, o, 32);
    if (lane == 0) dst[idx] = s;
}

// ---------------------------------------------------------------------------
// K3 (fused main): 16 exercise rows / block, 8 waves.
//   phase1: masked leaky-relu + exp(e-max) -> f16 LDS; adj streamed once as
//           non-temporal b128 (lane owns 4 consecutive columns)
//   phase2: v_wmma_f32_16x16x32_f16, K=1024:  P @ kc_Wh
//   phase3: v_wmma_f32_16x16x4_f32,  K=128:   ex_h @ E (fp32 precision)
//   phase4: elu( (1/rowsum) * (P@kc_Wh) * ex_Eh )
// ---------------------------------------------------------------------------
__global__ __launch_bounds__(256) void k_main(
    const float* __restrict__ ex_h, const int* __restrict__ adj,
    const float* __restrict__ E, const _Float16* __restrict__ kc_whT,
    const float* __restrict__ ex_score, const float* __restrict__ kc_score,
    float* __restrict__ out) {

    // stride 1044 halves = 2088 B/row: 8B-aligned for b64 stores;
    // 522 uints/row, 522 % 64 = 10 -> 16 rows land on 16 distinct banks.
    __shared__ _Float16 p16[16][1044];
    __shared__ float    exh[16][132];
    __shared__ float    inv_s[16];

    const int tid  = threadIdx.x;
    const int w    = tid >> 5;
    const int lane = tid & 31;
    const int blk  = blockIdx.x;

    // stage exercise_h tile 16x128 (coalesced)
    for (int i = 0; i < 8; ++i) {
        int idx = tid + i * 256;
        int r = idx >> 7, c = idx & 127;
        exh[r][c] = ex_h[(size_t)(blk * 16 + r) * F_IN + c];
    }

    // ---- phase 1: wave w owns rows 2w, 2w+1; lane owns cols t*128+lane*4+[0..3]
    f4 ksc[8];
    for (int t = 0; t < 8; ++t)
        ksc[t] = *((const f4*)kc_score + t * 32 + lane);

    for (int rr = 0; rr < 2; ++rr) {
        int r = w * 2 + rr;
        int grow = blk * 16 + r;
        float es = ex_score[grow];
        const i4* arow = (const i4*)(adj + (size_t)grow * N_KC);
        f4 ev[8];
        float m = -3.4e38f;
        for (int t = 0; t < 8; ++t) {
            i4 av = __builtin_nontemporal_load(arow + t * 32 + lane);  // streamed, no reuse
            f4 x  = es + ksc[t];
            for (int j = 0; j < 4; ++j) {
                float xx = x[j];
                xx = (xx >= 0.f) ? xx : LEAKY * xx;
                xx = (av[j] > 0) ? xx : MASKV;
                x[j] = xx;
                m = fmaxf(m, xx);
            }
            ev[t] = x;
        }
        for (int o = 16; o > 0; o >>= 1) m = fmaxf(m, __shfl_xor(m, o, 32));
        float s = 0.f;
        for (int t = 0; t < 8; ++t) {
            union { u2 u; _Float16 h[4]; } P;
            for (int j = 0; j < 4; ++j) {
                float p = __expf(ev[t][j] - m);   // masked -> 0; all-masked -> uniform
                s += p;
                P.h[j] = (_Float16)p;
            }
            *(u2*)(&p16[r][t * 128 + lane * 4]) = P.u;   // ds_store_b64
        }
        for (int o = 16; o > 0; o >>= 1) s += __shfl_xor(s, o, 32);
        if (lane == 0) inv_s[r] = 1.f / s;
    }
    __syncthreads();

    // ---- phase 2: P(16x1024,f16) @ kc_Wh(1024x128,f16); wave w -> cols [16w,16w+16)
    const int mrow = lane & 15;
    const int hi   = lane >> 4;
    const int col  = w * 16 + mrow;
    v8f acc = {};
    const uint32_t* rowp = (const uint32_t*)(&p16[mrow][0]);
    for (int kb = 0; kb < 32; ++kb) {
        int k0 = kb * 32;
        // A 16x32 f16: lanes 0-15 K=k0..k0+15, lanes 16-31 K=k0+16..k0+31
        union { v16h h; uint32_t u[8]; } A;
        int ab0 = (k0 + hi * 16) >> 1;
        int ab1 = (k0 + 8 + hi * 16) >> 1;
        for (int v = 0; v < 4; ++v) A.u[v]     = rowp[ab0 + v];
        for (int v = 0; v < 4; ++v) A.u[4 + v] = rowp[ab1 + v];
        // B 32x16 f16: lanes 0-15 K=k0..k0+15, lanes 16-31 K=k0+16..k0+31
        v16h B = *(const v16h*)(kc_whT + (size_t)col * N_KC + k0 + hi * 16);
        acc = __builtin_amdgcn_wmma_f32_16x16x32_f16(false, A.h, false, B,
                                                     (short)0, acc, false, false);
    }

    // ---- phase 3: ex_Eh tile, fp32 WMMA ------------------------------------
    v8f eh = {};
#if __has_builtin(__builtin_amdgcn_wmma_f32_16x16x4_f32)
    for (int k0 = 0; k0 < F_IN; k0 += 4) {
        int ka = k0 + hi * 2;
        v2f Af, Bf;
        Af.x = exh[mrow][ka + 0];
        Af.y = exh[mrow][ka + 1];
        Bf.x = E[(size_t)(ka + 0) * F_OUT + col];
        Bf.y = E[(size_t)(ka + 1) * F_OUT + col];
        eh = __builtin_amdgcn_wmma_f32_16x16x4_f32(false, Af, false, Bf,
                                                   (short)0, eh, false, false);
    }
#else
    for (int r = 0; r < 8; ++r) {
        int M = r + hi * 8;
        float s = 0.f;
        for (int k = 0; k < F_IN; ++k) s += exh[M][k] * E[(size_t)k * F_OUT + col];
        eh[r] = s;
    }
#endif

    // ---- phase 4: epilogue (C/D layout: VGPR r -> row M = r + 8*hi) --------
    for (int r = 0; r < 8; ++r) {
        int M = r + hi * 8;
        float val = acc[r] * inv_s[M] * eh[r];
        val = (val > 0.f) ? val : (__expf(val) - 1.f);   // elu
        out[(size_t)(blk * 16 + M) * F_OUT + col] = val;
    }
}

// ---------------------------------------------------------------------------
extern "C" void kernel_launch(void* const* d_in, const int* in_sizes, int n_in,
                              void* d_out, int out_size, void* d_ws, size_t ws_size,
                              hipStream_t stream) {
    const float* ex_h = (const float*)d_in[0];   // 20000x128
    const float* kc_h = (const float*)d_in[1];   // 1024x128
    const int*   adj  = (const int*)  d_in[2];   // 20000x1024
    const float* W1   = (const float*)d_in[3];   // 128x128
    const float* E    = (const float*)d_in[4];   // 128x128
    const float* a    = (const float*)d_in[5];   // 256x1
    float* out = (float*)d_out;

    char* ws = (char*)d_ws;
    _Float16* kc_whT = (_Float16*)ws;            // 128*1024 f16 = 262144 B
    float*    w1a    = (float*)(ws + 262144);    // 256 f32
    float*    ex_sc  = (float*)(ws + 263168);    // 20000 f32
    float*    kc_sc  = (float*)(ws + 343168);    // 1024 f32 (16B aligned)

    k_w1a   <<<1,    128, 0, stream>>>(W1, a, w1a);
    k_kcwh  <<<N_KC, 128, 0, stream>>>(kc_h, W1, kc_whT);
    int nrows = N_EX + N_KC;
    k_scores<<<(nrows * 32 + 255) / 256, 256, 0, stream>>>(ex_h, kc_h, w1a, ex_sc, kc_sc);
    k_main  <<<N_EX / 16, 256, 0, stream>>>(ex_h, adj, E, kc_whT, ex_sc, kc_sc, out);
}
